// RNN_26740466385734
// MI455X (gfx1250) — matrix operand order repaired and checked by reading, not compiled, not run on hardware
//
#include <hip/hip_runtime.h>
#include <hip/hip_bf16.h>
#include <math.h>

// ---------------------------------------------------------------------------
// Vanilla tanh-RNN for MI455X (gfx1250, wave32, WMMA, TDM).
//   Phase 1: xW = x @ W_ih^T + b_ih  -> written straight into d_out [T,B,H]
//            (bandwidth bound: ~200MB @ 23.3 TB/s ~ 9us; bf16 WMMA).
//   Phase 2: persistent scan kernel, 64 resident blocks. W_hh staged in LDS
//            ONCE via the Tensor Data Mover, then 512 sequential steps with a
//            device-wide generation barrier (removes 511 launches and 511
//            32KB-per-block LDS re-stagings from the serial critical path).
// bf16 operands / f32 accumulation via v_wmma_f32_16x16x32_bf16; tanh in f32.
// ---------------------------------------------------------------------------

typedef __attribute__((ext_vector_type(16))) __bf16 v16bf;
typedef __attribute__((ext_vector_type(8)))  float  v8f;
typedef __attribute__((ext_vector_type(4)))  unsigned int v4u_;
typedef __attribute__((ext_vector_type(8)))  int v8i_;
typedef __attribute__((ext_vector_type(4)))  int v4i_;

namespace {
constexpr int Tt = 512;   // timesteps
constexpr int Bb = 64;    // batch
constexpr int Ii = 512;   // input size
constexpr int Hh = 1024;  // hidden size
}

// ---- WMMA fragment loaders (CDNA5 wave32 16x16x32 bf16 layouts) ------------
// A (16xK=32, 16-bit): lanes 0-15 hold M=lane with K={k0+0..7, k0+16..23};
//                      lanes 16-31 hold M=lane-16 with K={k0+8..15, k0+24..31}.
__device__ inline v16bf load_a_f32(const float* __restrict__ p0, int ld,
                                   int lane, int k0) {
  const float* p = p0 + (size_t)(lane & 15) * ld + k0 + ((lane >> 4) << 3);
  v16bf a;
#pragma unroll
  for (int j = 0; j < 8; ++j) a[j] = (__bf16)p[j];
#pragma unroll
  for (int j = 0; j < 8; ++j) a[j + 8] = (__bf16)p[j + 16];
  return a;
}

__device__ inline v16bf load_a_bf16(const __bf16* __restrict__ p0, int ld,
                                    int lane, int k0) {
  const __bf16* p = p0 + (size_t)(lane & 15) * ld + k0 + ((lane >> 4) << 3);
  v16bf a;
#pragma unroll
  for (int j = 0; j < 8; ++j) a[j] = p[j];
#pragma unroll
  for (int j = 0; j < 8; ++j) a[j + 8] = p[j + 16];
  return a;
}

// B (K=32 x 16, 16-bit) staged in LDS as [k][n], n=0..15:
// lanes hold N=lane&15; lanes 0-15 hold K={k0+0..7, k0+16..23},
// lanes 16-31 hold K={k0+8..15, k0+24..31}.
__device__ inline v16bf load_b_lds(const __bf16* bs, int lane, int k0) {
  const __bf16* p = bs + (size_t)(k0 + ((lane >> 4) << 3)) * 16 + (lane & 15);
  v16bf b;
#pragma unroll
  for (int j = 0; j < 8; ++j) b[j] = p[j * 16];
#pragma unroll
  for (int j = 0; j < 8; ++j) b[j + 8] = p[(j + 16) * 16];
  return b;
}

// ---- device-wide generation barrier (persistent scan) ----------------------
__device__ inline void grid_barrier(unsigned* cnt, unsigned* gen) {
  __threadfence();   // release this block's global stores to agent scope
  __syncthreads();
  if (threadIdx.x == 0) {
    // capture generation BEFORE arriving (avoids missed-update deadlock)
    unsigned g = __hip_atomic_load(gen, __ATOMIC_RELAXED, __HIP_MEMORY_SCOPE_AGENT);
    unsigned prev = __hip_atomic_fetch_add(cnt, 1u, __ATOMIC_ACQ_REL,
                                           __HIP_MEMORY_SCOPE_AGENT);
    if (prev == gridDim.x - 1) {
      __hip_atomic_store(cnt, 0u, __ATOMIC_RELAXED, __HIP_MEMORY_SCOPE_AGENT);
      __hip_atomic_fetch_add(gen, 1u, __ATOMIC_ACQ_REL, __HIP_MEMORY_SCOPE_AGENT);
    } else {
      while (__hip_atomic_load(gen, __ATOMIC_ACQUIRE,
                               __HIP_MEMORY_SCOPE_AGENT) == g)
        __builtin_amdgcn_s_sleep(1);
    }
  }
  __syncthreads();
}

// ---- one-time weight transpose + downconvert + barrier reset ---------------
// BI[k*Hh + n] = bf16(W_ih[n*Ii + k]); BH[k*Hh + n] = bf16(W_hh[n*Hh + k]);
// hb[b*Hh + n] = bf16(h0[b*Hh + n]).
__global__ void rnn_convert_kernel(const float* __restrict__ Wih,
                                   const float* __restrict__ Whh,
                                   const float* __restrict__ h0,
                                   __bf16* __restrict__ BI,
                                   __bf16* __restrict__ BH,
                                   __bf16* __restrict__ hb,
                                   unsigned* __restrict__ barCnt,
                                   unsigned* __restrict__ barGen) {
  if (blockIdx.x == 0 && threadIdx.x == 0) { *barCnt = 0u; *barGen = 0u; }
  int idx = blockIdx.x * blockDim.x + threadIdx.x;
  int stride = gridDim.x * blockDim.x;
  for (int i = idx; i < Ii * Hh; i += stride) {
    int k = i / Hh, n = i - k * Hh;
    BI[i] = (__bf16)Wih[(size_t)n * Ii + k];
  }
  for (int i = idx; i < Hh * Hh; i += stride) {
    int k = i / Hh, n = i - k * Hh;
    BH[i] = (__bf16)Whh[(size_t)n * Hh + k];
  }
  for (int i = idx; i < Bb * Hh; i += stride) hb[i] = (__bf16)h0[i];
}

// ---- phase 1: input projection, out[m*Hh+n] = x[m,:].W_ih[n,:] + b_ih[n] ---
// grid: (Hh/16, (Tt*Bb)/128); block: 256 threads = 8 waves.
// Block stages a [Ii x 16] W_ih slab (16KB) in LDS, shared by 8 M-tiles.
__global__ void __launch_bounds__(256)
rnn_xproj_kernel(const float* __restrict__ x, const __bf16* __restrict__ BI,
                 const float* __restrict__ bih, float* __restrict__ out) {
  __shared__ unsigned int BsU[(Ii * 16) / 2];  // 16 KB
  const __bf16* Bs = (const __bf16*)BsU;
  const int colBase = blockIdx.x * 16;
  const int mBase = blockIdx.y * 128;
  const int lane = threadIdx.x & 31;
  const int wave = threadIdx.x >> 5;

  for (int u = threadIdx.x; u < (Ii * 16) / 2; u += 256) {
    int k = u >> 3, w = u & 7;
    BsU[u] = ((const unsigned int*)(BI + (size_t)k * Hh + colBase))[w];
  }
  __syncthreads();

  const int tileM = mBase + wave * 16;
  const float* arow = x + (size_t)tileM * Ii;  // rows m = t*Bb + b, ld = Ii
  const float bias = bih[colBase + (lane & 15)];
  v8f c;
#pragma unroll
  for (int r = 0; r < 8; ++r) c[r] = bias;

#pragma unroll 4
  for (int k0 = 0; k0 < Ii; k0 += 32) {
    if (k0 + 32 < Ii)
      __builtin_prefetch(arow + (size_t)(lane & 15) * Ii + k0 + 32, 0, 1);
    v16bf a = load_a_f32(arow, Ii, lane, k0);
    v16bf b = load_b_lds(Bs, lane, k0);
    c = __builtin_amdgcn_wmma_f32_16x16x32_bf16(false, a, false, b, (short)0,
                                                c, false, false);
  }

  // C/D layout: n = colBase + (lane&15); m = tileM + r + 8*(lane>=16)
  const int n = colBase + (lane & 15);
  const int m0 = tileM + ((lane >> 4) << 3);
#pragma unroll
  for (int r = 0; r < 8; ++r) out[(size_t)(m0 + r) * Hh + n] = c[r];
}

// ---- phase 2: persistent recurrence scan -----------------------------------
// 64 blocks x 128 threads (4 waves). Block owns a 16-wide N-slab: W_hh slab
// [Hh x 16] bf16 = 32KB staged in LDS ONCE via the Tensor Data Mover, then
// 512 steps separated by device-wide barriers. Each wave owns one 16x16 tile.
__global__ void __launch_bounds__(128)
rnn_scan_kernel(float* __restrict__ out,            // [T,B,H] (xW in, hs out)
                __bf16* __restrict__ hb0,           // ping (holds bf16(h0))
                __bf16* __restrict__ hb1,           // pong
                const __bf16* __restrict__ BH,      // [K=Hh][N=Hh] bf16
                const float* __restrict__ bhh,
                unsigned* __restrict__ barCnt, unsigned* __restrict__ barGen) {
  __shared__ unsigned int BsU[(Hh * 16) / 2];  // 32 KB W_hh N-slab
  const __bf16* Bs = (const __bf16*)BsU;
  const int colBase = blockIdx.x * 16;
  const int lane = threadIdx.x & 31;
  const int wave = threadIdx.x >> 5;

  // ---- one-time TDM stage: BH[k][colBase..colBase+16) -> LDS [k][0..16) ----
  if (wave == 0) {
    unsigned long long ga =
        (unsigned long long)(const void*)(BH + (size_t)colBase);
    unsigned ldsoff = (unsigned)(size_t)(void*)&BsU[0];  // low 32b = LDS offset
    v4u_ g0;
    g0[0] = 1u;                                    // count=1, user descriptor
    g0[1] = ldsoff;                                // lds_addr (bytes)
    g0[2] = (unsigned)(ga & 0xFFFFFFFFu);          // global_addr[31:0]
    g0[3] = (unsigned)((ga >> 32) & 0x01FFFFFFu)   // global_addr[56:32]
            | (2u << 30);                          // type=2 ("image")
    v8i_ g1;
    g1[0] = (int)(1u << 16);                // data_size=1 -> 2-byte elements
    g1[1] = (int)(16u << 16);               // tensor_dim0 = 16 (row length)
    g1[2] = (int)((unsigned)Hh << 16);      // tensor_dim1 = 1024 (rows)
    g1[3] = (int)(16u << 16);               // tile_dim0 = 16
    g1[4] = (int)Hh;                        // tile_dim1 = 1024, tile_dim2 = 0
    g1[5] = (int)Hh;                        // tensor_dim0_stride = 1024 elems
    g1[6] = 0;                              // stride hi / dim1_stride = 0
    g1[7] = 0;
    v4i_ z4 = {};
#if defined(__clang_major__) && (__clang_major__ >= 23)
    v8i_ z8 = {};
    __builtin_amdgcn_tensor_load_to_lds(g0, g1, z4, z4, z8, 0);
#else
    __builtin_amdgcn_tensor_load_to_lds(g0, g1, z4, z4, 0);
#endif
    __builtin_amdgcn_s_wait_tensorcnt(0);
  }
  __syncthreads();

  const int tileM = wave * 16;                 // Bb=64 -> 4 tiles = 4 waves
  const int n = colBase + (lane & 15);
  const int m0 = tileM + ((lane >> 4) << 3);
  const float bias = bhh[n];
  float* xw0 = out + (size_t)m0 * Hh + n;      // this lane's C/D column
  const size_t stepOff = (size_t)Bb * Hh;

  for (int t = 0; t < Tt; ++t) {
    const __bf16* hp = (t & 1) ? hb1 : hb0;
    __bf16* hn = (t & 1) ? hb0 : hb1;
    float* xw = xw0 + (size_t)t * stepOff;

    v8f c;
#pragma unroll
    for (int r = 0; r < 8; ++r) c[r] = xw[(size_t)r * Hh] + bias;
    if (t + 1 < Tt) __builtin_prefetch(xw + stepOff, 0, 1);

    const __bf16* arow = hp + (size_t)tileM * Hh;
#pragma unroll 4
    for (int k0 = 0; k0 < Hh; k0 += 32) {
      v16bf a = load_a_bf16(arow, Hh, lane, k0);
      v16bf b = load_b_lds(Bs, lane, k0);
      c = __builtin_amdgcn_wmma_f32_16x16x32_bf16(false, a, false, b, (short)0,
                                                  c, false, false);
    }

#pragma unroll
    for (int r = 0; r < 8; ++r) {
      float h = tanhf(c[r]);
      xw[(size_t)r * Hh] = h;                          // f32 output hs[t]
      hn[(size_t)(m0 + r) * Hh + n] = (__bf16)h;       // bf16 operand for t+1
    }

    grid_barrier(barCnt, barGen);  // all h_t visible before step t+1
  }
}

// ---------------------------------------------------------------------------
extern "C" void kernel_launch(void* const* d_in, const int* in_sizes, int n_in,
                              void* d_out, int out_size, void* d_ws,
                              size_t ws_size, hipStream_t stream) {
  const float* x    = (const float*)d_in[0];  // [T,B,I]
  const float* h0   = (const float*)d_in[1];  // [B,H]
  const float* W_ih = (const float*)d_in[2];  // [H,I]
  const float* b_ih = (const float*)d_in[3];  // [H]
  const float* W_hh = (const float*)d_in[4];  // [H,H]
  const float* b_hh = (const float*)d_in[5];  // [H]
  float* out = (float*)d_out;                 // [T,B,H]

  // workspace: 1MB BI + 2MB BH + 2 x 128KB h ping-pong + barrier words
  char* ws = (char*)d_ws;
  __bf16* BI  = (__bf16*)ws;
  __bf16* BH  = (__bf16*)(ws + (size_t)Ii * Hh * 2);
  __bf16* hb0 = (__bf16*)(ws + (size_t)Ii * Hh * 2 + (size_t)Hh * Hh * 2);
  __bf16* hb1 = hb0 + (size_t)Bb * Hh;
  size_t barOff = (size_t)Ii * Hh * 2 + (size_t)Hh * Hh * 2 +
                  2 * (size_t)Bb * Hh * 2;
  unsigned* barCnt = (unsigned*)(ws + barOff);
  unsigned* barGen = (unsigned*)(ws + barOff + 256);  // separate cachelines

  rnn_convert_kernel<<<2048, 256, 0, stream>>>(W_ih, W_hh, h0, BI, BH, hb0,
                                               barCnt, barGen);

  dim3 gproj(Hh / 16, (Tt * Bb) / 128);
  rnn_xproj_kernel<<<gproj, 256, 0, stream>>>(x, BI, b_ih, out);

  // persistent scan: 64 blocks (one per 16-wide N-slab), all co-resident
  // (32KB LDS each vs 320KB/WGP), 512 grid-barrier-separated steps.
  rnn_scan_kernel<<<Hh / 16, 128, 0, stream>>>(out, hb0, hb1, BH, b_hh,
                                               barCnt, barGen);
}
